// atae_LSTM_8796093022991
// MI455X (gfx1250) — compile-verified, matrix operand order checked by reading, not run on hardware
//
#include <hip/hip_runtime.h>
#include <hip/hip_bf16.h>

// ---------------------------------------------------------------------------
// ATAE-LSTM forward for MI455X (gfx1250, wave32, WMMA).
// Pipeline:
//   1. prep_convert : W_ih/W_hh f32->bf16, bsum = b_ih + b_hh
//   2. aspect_kernel: asp_emb (also written to d_out tail)
//   3. embed_kernel : masked [word_emb | asp_emb] -> bf16 A matrix (32768x1024)
//   4. xproj_kernel : A @ W_ih^T + bsum  via v_wmma_f32_16x16x32_bf16
//   5. lstm_kernel  : persistent, 512 sequential steps, W_hh slice in LDS,
//                     4-way split-K WMMA chains, next-step x_proj prefetch,
//                     grid barrier via phased atomic counter
// ---------------------------------------------------------------------------

#define Vv 50000
#define Dd 512
#define Bb 64
#define Ll 512
#define Aa 4
#define G4 2048   /* 4*D  */
#define G2 1024   /* 2*D  */

typedef __attribute__((ext_vector_type(16))) __bf16        v16bf;
typedef __attribute__((ext_vector_type(8)))  float         v8f;
typedef __attribute__((ext_vector_type(4)))  unsigned int  v4u;

union Frag16 { v16bf v; v4u q[2]; };

__device__ __forceinline__ unsigned short f32_bf16(float f) {
  unsigned u = __float_as_uint(f);
  unsigned r = u + 0x7FFFu + ((u >> 16) & 1u);   // round-to-nearest-even
  return (unsigned short)(r >> 16);
}

__device__ __forceinline__ float sigmoidf(float x) {
  return 1.0f / (1.0f + __expf(-x));
}

__device__ __forceinline__ v8f wmma_bf16(v16bf a, v16bf b, v8f c) {
  // (neg_a, A, neg_b, B, c_mod, C, reuse_a, reuse_b)
  return __builtin_amdgcn_wmma_f32_16x16x32_bf16(false, a, false, b,
                                                 (short)0, c, false, false);
}

// Device-wide barrier: monotonically increasing phase targets; counter is
// zeroed with hipMemsetAsync before the persistent kernel launches.
__device__ __forceinline__ void gbar(unsigned* bar, unsigned nwg, unsigned phase) {
  __syncthreads();
  if (threadIdx.x == 0) {
    __threadfence();
    __hip_atomic_fetch_add(bar, 1u, __ATOMIC_ACQ_REL, __HIP_MEMORY_SCOPE_AGENT);
    while (__hip_atomic_load(bar, __ATOMIC_ACQUIRE, __HIP_MEMORY_SCOPE_AGENT)
           < nwg * phase) {
      __builtin_amdgcn_s_sleep(1);
    }
  }
  __syncthreads();
}

// ---------------------------------------------------------------------------
// 1. Convert weights to bf16, build summed bias.
// ---------------------------------------------------------------------------
__global__ void prep_convert(const float* __restrict__ Wih,
                             const float* __restrict__ Whh,
                             const float* __restrict__ bih,
                             const float* __restrict__ bhh,
                             unsigned short* __restrict__ wih_b,
                             unsigned short* __restrict__ whh_b,
                             float* __restrict__ bsum) {
  const int N1 = G4 * G2;        // 2048*1024
  const int N2 = G4 * Dd;        // 2048*512
  const int total = N1 + N2 + G4;
  for (int i = blockIdx.x * blockDim.x + threadIdx.x; i < total;
       i += gridDim.x * blockDim.x) {
    if (i < N1) {
      wih_b[i] = f32_bf16(Wih[i]);
    } else if (i < N1 + N2) {
      whh_b[i - N1] = f32_bf16(Whh[i - N1]);
    } else {
      int j = i - N1 - N2;
      bsum[j] = bih[j] + bhh[j];
    }
  }
}

// ---------------------------------------------------------------------------
// 2. Aspect embedding mean. Writes ws copy + final output region.
// ---------------------------------------------------------------------------
__global__ void aspect_kernel(const int* __restrict__ aspect,
                              const int* __restrict__ asplen,
                              const float* __restrict__ emb,
                              float* __restrict__ asp_ws,
                              float* __restrict__ asp_out) {
  int i = blockIdx.x * blockDim.x + threadIdx.x;
  if (i >= Bb * Dd) return;
  int b = i >> 9, d = i & (Dd - 1);
  float s = 0.0f;
  for (int a = 0; a < Aa; ++a)
    s += emb[(size_t)aspect[b * Aa + a] * Dd + d];
  s /= (float)asplen[b];
  asp_ws[i]  = s;
  asp_out[i] = s;
}

// ---------------------------------------------------------------------------
// 3. Build masked bf16 A matrix: rows = b*L + t, cols = [word(0:512)|asp(512:1024)]
// ---------------------------------------------------------------------------
__global__ void embed_kernel(const int* __restrict__ X,
                             const int* __restrict__ Xlen,
                             const float* __restrict__ emb,
                             const float* __restrict__ asp_ws,
                             unsigned short* __restrict__ Ab) {
  const int total = Bb * Ll * G2;  // 33,554,432
  for (int i = blockIdx.x * blockDim.x + threadIdx.x; i < total;
       i += gridDim.x * blockDim.x) {
    int row = i >> 10, col = i & (G2 - 1);
    int b = row >> 9, t = row & (Ll - 1);
    float v = 0.0f;
    if (t < Xlen[b]) {
      v = (col < Dd) ? emb[(size_t)X[row] * Dd + col]
                     : asp_ws[b * Dd + (col - Dd)];
    }
    Ab[i] = f32_bf16(v);
  }
}

// ---------------------------------------------------------------------------
// 4. x_proj GEMM: (32768 x 1024) * (1024 x 2048)^T + bsum.
//    Per WG: stage 16x1024 A-strip in LDS (32 KB), 8 waves each do a 16x128
//    output strip (8 accumulator tiles) reusing one A fragment per k-step.
// ---------------------------------------------------------------------------
__global__ void __launch_bounds__(256) xproj_kernel(
    const unsigned short* __restrict__ Ab,
    const unsigned short* __restrict__ Wb,
    const float* __restrict__ bsum,
    float* __restrict__ xproj) {
  __shared__ unsigned short sA[16 * G2];   // 32 KB
  const int tid = threadIdx.x;
  const int mstrip = blockIdx.x >> 1;      // 0..2047
  const int m0 = mstrip * 16;

  {  // cooperative stage of A rows m0..m0+15 (contiguous 32 KB)
    const v4u* src = (const v4u*)(Ab + (size_t)m0 * G2);
    v4u* dst = (v4u*)sA;
    for (int i = tid; i < (16 * G2) / 8; i += 256) dst[i] = src[i];
  }
  __syncthreads();

  const int lane = tid & 31;
  const int wv   = tid >> 5;
  const int nstrip = (blockIdx.x & 1) * 8 + wv;   // 0..15
  const int n0 = nstrip * 128;
  const int half = lane >> 4;
  const int mrow = lane & 15;
  const int ncol = lane & 15;

  v8f acc[8];
  for (int j = 0; j < 8; ++j) {
    float bv = bsum[n0 + j * 16 + ncol];
#pragma unroll
    for (int i = 0; i < 8; ++i) acc[j][i] = bv;
  }

  const unsigned short* Ar = sA + mrow * G2;
  for (int k = 0; k < G2; k += 32) {
    Frag16 a;
    a.q[0] = *(const v4u*)(Ar + k + half * 8);        // ds_load_b128
    a.q[1] = *(const v4u*)(Ar + k + 16 + half * 8);
#pragma unroll
    for (int j = 0; j < 8; ++j) {
      const unsigned short* Br = Wb + (size_t)(n0 + j * 16 + ncol) * G2 + k;
      Frag16 bb;
      bb.q[0] = *(const v4u*)(Br + half * 8);         // global_load_b128
      bb.q[1] = *(const v4u*)(Br + 16 + half * 8);
      acc[j] = wmma_bf16(a.v, bb.v, acc[j]);
    }
  }

  for (int j = 0; j < 8; ++j) {
#pragma unroll
    for (int i = 0; i < 8; ++i) {
      xproj[(size_t)(m0 + i + 8 * half) * G4 + n0 + j * 16 + ncol] = acc[j][i];
    }
  }
}

// ---------------------------------------------------------------------------
// 5. Persistent LSTM. 64 WGs x 8 waves = 512 waves = 4 Mtile x 128 Ntile per
//    step. Each WG pins its 32-col W_hh slice (32 KB bf16) in LDS for all 512
//    steps. Per-tile GEMM uses 4 independent split-K WMMA chains to cut the
//    serial WMMA->WMMA hazard latency on the recurrence critical path, and
//    speculatively prefetches step t+1's x_proj slice (global_prefetch_b8).
// ---------------------------------------------------------------------------
__global__ void __launch_bounds__(256) lstm_kernel(
    const float* __restrict__ xproj,
    const unsigned short* __restrict__ Whb,
    const int* __restrict__ Xlen,
    float* __restrict__ h, float* __restrict__ c,
    unsigned short* __restrict__ hb,
    float* __restrict__ gates,
    unsigned* __restrict__ bar,
    float* __restrict__ out, float* __restrict__ hid) {
  __shared__ unsigned short sB[32 * Dd];   // 32 KB: W_hh rows for this WG
  const int tid = threadIdx.x;
  const int n0g = blockIdx.x * 32;         // this WG's 32 gate columns

  {  // stage W_hh slice once (rows n0g..n0g+31 are contiguous, 32 KB)
    const v4u* src = (const v4u*)(Whb + (size_t)n0g * Dd);
    v4u* dst = (v4u*)sB;
    for (int i = tid; i < (32 * Dd) / 8; i += 256) dst[i] = src[i];
  }
  __syncthreads();

  const int lane  = tid & 31;
  const int wave  = blockIdx.x * 8 + (tid >> 5);  // 0..511
  const int mtile = wave & 3;                     // batch tile (x16)
  const int ntile = wave >> 2;                    // gate-col tile (x16)
  const int half  = lane >> 4;
  const int mrow  = lane & 15;
  const int ncol  = lane & 15;
  const int ngcol = ntile * 16 + ncol;            // global gate col
  const int nlcol = ngcol - n0g;                  // 0..31 (LDS row)
  const int myb   = mtile * 16;
  const int tg    = blockIdx.x * 256 + tid;       // 0..16383
  const unsigned nwg = gridDim.x;

  for (int t = 0; t < Ll; ++t) {
    // --- phase A: gates = x_proj[:,t,:] + h @ W_hh^T -----------------------
    v8f acc0, acc1, acc2, acc3;
#pragma unroll
    for (int i = 0; i < 8; ++i) {
      acc0[i] = xproj[((size_t)(myb + i + 8 * half) * Ll + t) * G4 + ngcol];
      acc1[i] = 0.0f;
      acc2[i] = 0.0f;
      acc3[i] = 0.0f;
    }
    if (t + 1 < Ll) {
      // hide next step's scattered f32 load latency behind this step's WMMAs
#pragma unroll
      for (int i = 0; i < 8; ++i)
        __builtin_prefetch(
            &xproj[((size_t)(myb + i + 8 * half) * Ll + (t + 1)) * G4 + ngcol],
            0, 3);
    }

    const unsigned short* Ar = hb + (myb + mrow) * Dd;
    const unsigned short* Br = sB + nlcol * Dd;
#pragma unroll
    for (int k = 0; k < Dd; k += 32) {
      Frag16 a, bb;
      a.q[0]  = *(const v4u*)(Ar + k + half * 8);       // h (bf16, L2)
      a.q[1]  = *(const v4u*)(Ar + k + 16 + half * 8);
      bb.q[0] = *(const v4u*)(Br + k + half * 8);       // ds_load_b128
      bb.q[1] = *(const v4u*)(Br + k + 16 + half * 8);
      switch ((k >> 5) & 3) {                           // 4 independent chains
        case 0: acc0 = wmma_bf16(a.v, bb.v, acc0); break;
        case 1: acc1 = wmma_bf16(a.v, bb.v, acc1); break;
        case 2: acc2 = wmma_bf16(a.v, bb.v, acc2); break;
        default: acc3 = wmma_bf16(a.v, bb.v, acc3); break;
      }
    }
    v8f accs = (acc0 + acc1) + (acc2 + acc3);
#pragma unroll
    for (int i = 0; i < 8; ++i)
      gates[(myb + i + 8 * half) * G4 + ngcol] = accs[i];

    gbar(bar, nwg, 2u * (unsigned)t + 1u);

    // --- phase B: pointwise cell update, 2 hidden units per thread ---------
#pragma unroll
    for (int e = 0; e < 2; ++e) {
      int idx = tg + e * 16384;                  // 0..32767 -> (b, d)
      int b = idx >> 9, d = idx & (Dd - 1);
      const float* gr = gates + b * G4;
      float ii = sigmoidf(gr[d]);
      float ff = sigmoidf(gr[Dd + d]);
      float g_ = tanhf(gr[2 * Dd + d]);
      float oo = sigmoidf(gr[3 * Dd + d]);
      float cn = ff * c[idx] + ii * g_;
      float hn = oo * tanhf(cn);
      bool m = t < Xlen[b];
      float hc = m ? hn : h[idx];
      float cc = m ? cn : c[idx];
      h[idx] = hc;
      c[idx] = cc;
      hb[idx] = f32_bf16(hc);
      out[((size_t)b * Ll + t) * Dd + d] = m ? hn : 0.0f;
    }
    gbar(bar, nwg, 2u * (unsigned)t + 2u);
  }

  // final hidden state hT -> output[1] region
#pragma unroll
  for (int e = 0; e < 2; ++e) {
    int idx = tg + e * 16384;
    hid[idx] = h[idx];
  }
}

// ---------------------------------------------------------------------------
extern "C" void kernel_launch(void* const* d_in, const int* in_sizes, int n_in,
                              void* d_out, int out_size, void* d_ws, size_t ws_size,
                              hipStream_t stream) {
  (void)in_sizes; (void)n_in; (void)out_size;
  const int*   X      = (const int*)d_in[0];
  const int*   Xlen   = (const int*)d_in[1];
  const int*   aspect = (const int*)d_in[2];
  const int*   asplen = (const int*)d_in[3];
  const float* emb    = (const float*)d_in[4];
  const float* Wih    = (const float*)d_in[5];
  const float* Whh    = (const float*)d_in[6];
  const float* bih    = (const float*)d_in[7];
  const float* bhh    = (const float*)d_in[8];

  char* ws = (char*)d_ws;
  // workspace layout (bytes)
  size_t oA    = 0;                                      // A bf16: 32768x1024
  size_t oWih  = oA   + (size_t)Bb * Ll * G2 * 2;        // 64 MB
  size_t oWhh  = oWih + (size_t)G4 * G2 * 2;             // +4 MB
  size_t oX    = oWhh + (size_t)G4 * Dd * 2;             // +2 MB
  size_t oH    = oX   + (size_t)Bb * Ll * G4 * 4;        // +256 MB
  size_t oC    = oH   + (size_t)Bb * Dd * 4;
  size_t oHb   = oC   + (size_t)Bb * Dd * 4;
  size_t oG    = oHb  + (size_t)Bb * Dd * 2;
  size_t oAsp  = oG   + (size_t)Bb * G4 * 4;
  size_t oBs   = oAsp + (size_t)Bb * Dd * 4;
  size_t oBar  = oBs  + (size_t)G4 * 4;
  size_t need  = oBar + 256;
  if (ws_size < need) return;

  float* outF    = (float*)d_out;
  float* hid_out = outF + (size_t)Bb * Ll * Dd;
  float* asp_out = hid_out + (size_t)Bb * Dd;

  // zero h, c, h_bf16 and the barrier counter (graph-capturable)
  hipMemsetAsync(ws + oH,   0, oG - oH, stream);
  hipMemsetAsync(ws + oBar, 0, 256,     stream);

  prep_convert<<<2048, 256, 0, stream>>>(
      Wih, Whh, bih, bhh,
      (unsigned short*)(ws + oWih), (unsigned short*)(ws + oWhh),
      (float*)(ws + oBs));

  aspect_kernel<<<(Bb * Dd + 255) / 256, 256, 0, stream>>>(
      aspect, asplen, emb, (float*)(ws + oAsp), asp_out);

  embed_kernel<<<8192, 256, 0, stream>>>(
      X, Xlen, emb, (const float*)(ws + oAsp), (unsigned short*)(ws + oA));

  xproj_kernel<<<4096, 256, 0, stream>>>(
      (const unsigned short*)(ws + oA), (const unsigned short*)(ws + oWih),
      (const float*)(ws + oBs), (float*)(ws + oX));

  lstm_kernel<<<64, 256, 0, stream>>>(
      (const float*)(ws + oX), (const unsigned short*)(ws + oWhh), Xlen,
      (float*)(ws + oH), (float*)(ws + oC), (unsigned short*)(ws + oHb),
      (float*)(ws + oG), (unsigned*)(ws + oBar), outF, hid_out);
}
